// MPEmbeddingDecoder_75849122448086
// MI455X (gfx1250) — matrix operand order repaired and checked by reading, not compiled, not run on hardware
//
#include <hip/hip_runtime.h>
#include <hip/hip_bf16.h>
#include <math.h>

// ---------------- problem constants ----------------
#define Bn    32
#define Kq    100
#define Cc    256
#define HHn   8
#define Lv    4
#define Pn    4
#define LOUTn 3
#define DFFn  1024
#define NFn   128
#define DHn   32
#define Ntot  5440          // 64*64 + 32*32 + 16*16 + 8*8
#define BK    (Bn*Kq)       // 3200
#define BKC   ((size_t)BK*Cc)

// ---------------- WMMA types ----------------
typedef __bf16 v16bf __attribute__((ext_vector_type(16)));
typedef __bf16 v8bf  __attribute__((ext_vector_type(8)));
typedef float  v8f   __attribute__((ext_vector_type(8)));

__device__ __forceinline__ __bf16 f2bf(float x) {
    unsigned u = __builtin_bit_cast(unsigned, x);
    unsigned r = u + 0x7FFFu + ((u >> 16) & 1u);   // round-to-nearest-even
    unsigned short h = (unsigned short)(r >> 16);
    return __builtin_bit_cast(__bf16, h);
}

// ============================================================
// f32 -> bf16 conversion (memory-bound, done once per operand)
// ============================================================
__global__ void __launch_bounds__(256)
cvt_bf16_kernel(const float* __restrict__ a, __bf16* __restrict__ o, size_t n)
{
    size_t i = (size_t)blockIdx.x * 256 + threadIdx.x;
    if (i < n) o[i] = f2bf(a[i]);
}

// ============================================================
// WMMA GEMM:  C[M,N] = A[M,Kd] @ W[N,Kd]^T + bias   (optional relu)
// bf16 operands (pre-converted), f32 accumulate, fp32 out.
// One wave -> 32(M) x 64(N) block: 2 A-frags x 4 B-frags = 8 wmma
// per 32-K step, with B fragments reused across both M tiles.
// Requires M%32==0, Kd%32==0, N%64==0.
// ============================================================
__global__ void __launch_bounds__(256)
wmma_gemm_bf16(const __bf16* __restrict__ A, const __bf16* __restrict__ W,
               const float* __restrict__ bias, float* __restrict__ C,
               int M, int Kd, int N, int relu)
{
    const int wid    = (blockIdx.x * 256 + threadIdx.x) >> 5;
    const int lane   = threadIdx.x & 31;
    const int tilesN = N >> 6;
    const int tm     = (wid / tilesN) << 5;   // 32 rows per wave
    const int tn     = (wid % tilesN) << 6;   // 64 cols per wave
    if (tm >= M) return;

    const int half = lane >> 4;      // 0 or 1
    const int lm   = lane & 15;

    v8f acc[8];
    #pragma unroll
    for (int i = 0; i < 8; ++i) acc[i] = v8f{};

    const __bf16* arow0 = A + (size_t)(tm + lm) * Kd;        // M tile 0
    const __bf16* arow1 = A + (size_t)(tm + 16 + lm) * Kd;   // M tile 1

    #pragma unroll 2
    for (int kb = 0; kb < Kd; kb += 32) {
        // ---- A fragments: 16x32 bf16 (ISA layout) ----
        // lane L: M = L%16; elems 0-7 -> K = kb + (L/16)*8 + j ; elems 8-15 -> +16
        const __bf16* ap0 = arow0 + kb + half * 8;
        const __bf16* ap1 = arow1 + kb + half * 8;
        if (kb + 32 < Kd) {
            __builtin_prefetch(ap0 + 32, 0, 3);
            __builtin_prefetch(ap1 + 32, 0, 3);
        }
        v16bf af0 = __builtin_shufflevector(*(const v8bf*)ap0, *(const v8bf*)(ap0 + 16),
                      0,1,2,3,4,5,6,7,8,9,10,11,12,13,14,15);
        v16bf af1 = __builtin_shufflevector(*(const v8bf*)ap1, *(const v8bf*)(ap1 + 16),
                      0,1,2,3,4,5,6,7,8,9,10,11,12,13,14,15);

        // ---- 4 B fragments (32x16 each), loaded up front ----
        // lane L holds B[k][n]: elems 0-7 -> K = kb + (L/16)*8 + e ; elems 8-15 -> +16
        v16bf bf[4];
        #pragma unroll
        for (int t = 0; t < 4; ++t) {
            const __bf16* wp = W + (size_t)(tn + t * 16 + lm) * Kd + kb + half * 8;
            bf[t] = __builtin_shufflevector(*(const v8bf*)wp, *(const v8bf*)(wp + 16),
                      0,1,2,3,4,5,6,7,8,9,10,11,12,13,14,15);
        }

        // ---- 8 back-to-back WMMAs, B reused across both M tiles ----
        #pragma unroll
        for (int t = 0; t < 4; ++t)
            acc[t] = __builtin_amdgcn_wmma_f32_16x16x32_bf16(
                         false, af0, false, bf[t], (short)0, acc[t], false, false);
        #pragma unroll
        for (int t = 0; t < 4; ++t)
            acc[4 + t] = __builtin_amdgcn_wmma_f32_16x16x32_bf16(
                         false, af1, false, bf[t], (short)0, acc[4 + t], false, false);
    }

    // ---- store: C/D layout: VGPR r, lane L -> M = r + (L/16)*8, N = L%16 ----
    #pragma unroll
    for (int t = 0; t < 4; ++t) {
        const int n  = tn + t * 16 + lm;
        const float bv = bias ? bias[n] : 0.0f;
        #pragma unroll
        for (int mt = 0; mt < 2; ++mt) {
            #pragma unroll
            for (int r = 0; r < 8; ++r) {
                const int mm = tm + mt * 16 + half * 8 + r;
                float v = acc[mt * 4 + t][r] + bv;
                if (relu) v = v > 0.0f ? v : 0.0f;
                C[(size_t)mm * N + n] = v;
            }
        }
    }
}

// ============================================================
// build value[B,Ntot,Cc] from the four NCHW feature maps
// ============================================================
__global__ void build_value_kernel(const float* __restrict__ f0, const float* __restrict__ f1,
                                   const float* __restrict__ f2, const float* __restrict__ f3,
                                   float* __restrict__ value, __bf16* __restrict__ value_bf)
{
    size_t i = (size_t)blockIdx.x * 256 + threadIdx.x;
    if (i >= (size_t)Bn * Ntot * Cc) return;
    int c = (int)(i % Cc);
    size_t t = i / Cc;
    int n = (int)(t % Ntot);
    int b = (int)(t / Ntot);
    const float* f; int hw, start;
    if      (n < 4096) { f = f0; hw = 4096; start = 0;    }
    else if (n < 5120) { f = f1; hw = 1024; start = 4096; }
    else if (n < 5376) { f = f2; hw = 256;  start = 5120; }
    else               { f = f3; hw = 64;   start = 5376; }
    float v = f[(size_t)(b * Cc + c) * hw + (n - start)];
    value[i] = v;
    value_bf[i] = f2bf(v);
}

// ============================================================
// valid[b,k] = (max |query_embed[b,k,:]| != 0)
// ============================================================
__global__ void __launch_bounds__(256)
valid_kernel(const float* __restrict__ qe0, float* __restrict__ valid)
{
    int wid  = (blockIdx.x * 256 + threadIdx.x) >> 5;
    int lane = threadIdx.x & 31;
    if (wid >= BK) return;
    const float* r = qe0 + (size_t)wid * Cc;
    float m = 0.0f;
    for (int j = lane; j < Cc; j += 32) m = fmaxf(m, fabsf(r[j]));
    for (int o = 16; o > 0; o >>= 1) m = fmaxf(m, __shfl_xor(m, o));
    if (lane == 0) valid[wid] = (m != 0.0f) ? 1.0f : 0.0f;
}

// ============================================================
// sine positional encoding of pts -> qpos[B,K,256]
// ============================================================
__global__ void coord_pos_kernel(const float* __restrict__ pts, float* __restrict__ qpos)
{
    int i = blockIdx.x * 256 + threadIdx.x;
    if (i >= (int)BKC) return;
    int d   = i & 255;
    int row = i >> 8;
    float coord = (d < NFn) ? pts[row * 2 + 1] : pts[row * 2 + 0]; // first half: y, second: x
    int dd = d & (NFn - 1);
    int j  = dd >> 1;
    float t   = powf(10000.0f, (float)j / 64.0f);
    float ang = coord * 6.283185307179586f / t;
    qpos[i] = (dd & 1) ? cosf(ang) : sinf(ang);
}

// ============================================================
// elementwise add (fp32 out + bf16 mirror for the next GEMM)
// ============================================================
__global__ void add_vec_kernel(const float* __restrict__ a, const float* __restrict__ b,
                               float* __restrict__ o, __bf16* __restrict__ obf, int n)
{
    int i = blockIdx.x * 256 + threadIdx.x;
    if (i < n) {
        float v = a[i] + b[i];
        o[i] = v;
        obf[i] = f2bf(v);
    }
}

// ============================================================
// self-attention core: one wave per (b,h,q); 8 waves/block
// Q,K,V,O: [B,K,HH*DH] with head channel h*32+d
// ============================================================
__global__ void __launch_bounds__(256)
sa_attn_kernel(const float* __restrict__ Q, const float* __restrict__ Kt,
               const float* __restrict__ V, float* __restrict__ O)
{
    __shared__ float sc[8][Kq];
    int wline = threadIdx.x >> 5;
    int lane  = threadIdx.x & 31;
    int unit  = blockIdx.x * 8 + wline;       // (b,h,q)
    int q  = unit % Kq;
    int bh = unit / Kq;
    int h  = bh % HHn;
    int b  = bh / HHn;
    const float scale = 0.17677669529663687f; // 1/sqrt(32)
    const float* qrow = Q + (size_t)(b * Kq + q) * Cc + h * DHn;
    for (int kk = lane; kk < Kq; kk += 32) {
        const float* krow = Kt + (size_t)(b * Kq + kk) * Cc + h * DHn;
        float dot = 0.0f;
        #pragma unroll
        for (int d = 0; d < DHn; ++d) dot += qrow[d] * krow[d];
        sc[wline][kk] = dot * scale;
    }
    __syncthreads();
    float mx = -1e30f;
    for (int kk = 0; kk < Kq; ++kk) mx = fmaxf(mx, sc[wline][kk]);
    float Z = 0.0f;
    for (int kk = 0; kk < Kq; ++kk) Z += expf(sc[wline][kk] - mx);
    float inv = 1.0f / Z;
    int d = lane;
    float acc = 0.0f;
    for (int kk = 0; kk < Kq; ++kk) {
        float a = expf(sc[wline][kk] - mx) * inv;
        acc += a * V[(size_t)(b * Kq + kk) * Cc + h * DHn + d];
    }
    O[(size_t)(b * Kq + q) * Cc + h * DHn + d] = acc;
}

// ============================================================
// O = LayerNorm(X + R)*g + b   [* valid]   (one wave per row)
// also writes bf16 mirror of O when obf != nullptr
// ============================================================
__global__ void __launch_bounds__(256)
add_ln_kernel(const float* __restrict__ X, const float* __restrict__ R,
              const float* __restrict__ g, const float* __restrict__ bb,
              const float* __restrict__ valid, float* __restrict__ O,
              __bf16* __restrict__ obf, int useValid)
{
    int wid  = (blockIdx.x * 256 + threadIdx.x) >> 5;
    int lane = threadIdx.x & 31;
    if (wid >= BK) return;
    const float* x = X + (size_t)wid * Cc;
    const float* r = R + (size_t)wid * Cc;
    float v[8];
    float s = 0.0f;
    #pragma unroll
    for (int j = 0; j < 8; ++j) { v[j] = x[lane + j * 32] + r[lane + j * 32]; s += v[j]; }
    for (int o = 16; o > 0; o >>= 1) s += __shfl_xor(s, o);
    float mu = s * (1.0f / 256.0f);
    float qv = 0.0f;
    #pragma unroll
    for (int j = 0; j < 8; ++j) { float d = v[j] - mu; qv += d * d; }
    for (int o = 16; o > 0; o >>= 1) qv += __shfl_xor(qv, o);
    float rs = rsqrtf(qv * (1.0f / 256.0f) + 1e-5f);
    float vm = useValid ? valid[wid] : 1.0f;
    #pragma unroll
    for (int j = 0; j < 8; ++j) {
        int c = lane + j * 32;
        float ov = ((v[j] - mu) * rs * g[c] + bb[c]) * vm;
        O[(size_t)wid * Cc + c] = ov;
        if (obf) obf[(size_t)wid * Cc + c] = f2bf(ov);
    }
}

// ============================================================
// bilinear tap (grid_sample align_corners=False, zeros padding)
// ============================================================
__device__ __forceinline__ float tapv(const float* __restrict__ base, int start, int w, int h,
                                      float xi, float yi, int c)
{
    bool ok = (xi >= 0.0f) && (xi <= (float)(w - 1)) && (yi >= 0.0f) && (yi <= (float)(h - 1));
    int xc = (int)fminf(fmaxf(xi, 0.0f), (float)(w - 1));
    int yc = (int)fminf(fmaxf(yi, 0.0f), (float)(h - 1));
    float v = base[(size_t)(start + yc * w + xc) * Cc + c];
    return ok ? v : 0.0f;
}

__device__ __forceinline__ float bilin(const float* __restrict__ base, int start, int w, int h,
                                       float lx, float ly, int c)
{
    float x = lx * (float)w - 0.5f, y = ly * (float)h - 0.5f;
    float x0 = floorf(x), y0 = floorf(y);
    float wx = x - x0, wy = y - y0;
    float v00 = tapv(base, start, w, h, x0,        y0,        c);
    float v10 = tapv(base, start, w, h, x0 + 1.0f, y0,        c);
    float v01 = tapv(base, start, w, h, x0,        y0 + 1.0f, c);
    float v11 = tapv(base, start, w, h, x0 + 1.0f, y0 + 1.0f, c);
    return (v00 * (1.0f - wx) + v10 * wx) * (1.0f - wy)
         + (v01 * (1.0f - wx) + v11 * wx) * wy;
}

__constant__ int c_hs[4] = {64, 32, 16, 8};
__constant__ int c_st[4] = {0, 4096, 5120, 5376};

// ============================================================
// qe += mean over levels of bilinear-sampled raw value  (* valid)
// block per (b,k), thread per channel
// ============================================================
__global__ void __launch_bounds__(256)
ref_feat_kernel(const float* __restrict__ value, const float* __restrict__ pts,
                const float* __restrict__ valid, float* __restrict__ qe)
{
    int bk = blockIdx.x;
    int c  = threadIdx.x;
    int b  = bk / Kq;
    float px = pts[bk * 2 + 0], py = pts[bk * 2 + 1];
    const float* base = value + (size_t)b * Ntot * Cc;
    float acc = 0.0f;
    #pragma unroll
    for (int l = 0; l < 4; ++l)
        acc += bilin(base, c_st[l], c_hs[l], c_hs[l], px, py, c);
    qe[(size_t)bk * Cc + c] += acc * 0.25f * valid[bk];
}

// ============================================================
// deformable sampling + attention-weighted sum
// one wave per (b,k,h); lane = head channel d
// writes fp32 + bf16 mirror (input of output-projection GEMM)
// ============================================================
__global__ void __launch_bounds__(256)
deform_kernel(const float* __restrict__ vproj, const float* __restrict__ off,
              const float* __restrict__ awr, const float* __restrict__ pts,
              float* __restrict__ O, __bf16* __restrict__ obf)
{
    int wline = threadIdx.x >> 5;
    int lane  = threadIdx.x & 31;
    int unit  = blockIdx.x * 8 + wline;   // (b,k,h) with h fastest
    int h  = unit % HHn;
    int bk = unit / HHn;
    int b  = bk / Kq;

    const float* aw = awr + (size_t)bk * (HHn * Lv * Pn) + h * (Lv * Pn);
    float mx = -1e30f;
    #pragma unroll
    for (int j = 0; j < 16; ++j) mx = fmaxf(mx, aw[j]);
    float Z = 0.0f;
    #pragma unroll
    for (int j = 0; j < 16; ++j) Z += expf(aw[j] - mx);
    float inv = 1.0f / Z;

    float px = pts[bk * 2 + 0], py = pts[bk * 2 + 1];
    const float* offr = off + (size_t)bk * Cc + h * (Lv * Pn * 2);
    const float* base = vproj + (size_t)b * Ntot * Cc;
    int ch = h * DHn + lane;

    float acc = 0.0f;
    #pragma unroll
    for (int l = 0; l < 4; ++l) {
        int hw = c_hs[l];
        #pragma unroll
        for (int p = 0; p < 4; ++p) {
            int j = l * 4 + p;
            float w8 = expf(aw[j] - mx) * inv;
            float lx = px + offr[j * 2 + 0] / (float)hw;  // norm = (W,H)
            float ly = py + offr[j * 2 + 1] / (float)hw;
            acc += w8 * bilin(base, c_st[l], hw, hw, lx, ly, ch);
        }
    }
    O[(size_t)bk * Cc + ch] = acc;
    obf[(size_t)bk * Cc + ch] = f2bf(acc);
}

// ============================================================
// point head tail: delta = phh @ w2^T + b2 ; pts = sigmoid(inv_sigmoid(pts)+delta)
// one wave per (b,k)
// ============================================================
__global__ void __launch_bounds__(256)
point_head_kernel(const float* __restrict__ phh, const float* __restrict__ w2,
                  const float* __restrict__ b2, float* __restrict__ pts,
                  float* __restrict__ out_pts, int layer)
{
    int wid  = (blockIdx.x * 256 + threadIdx.x) >> 5;
    int lane = threadIdx.x & 31;
    if (wid >= BK) return;
    const float* r = phh + (size_t)wid * Cc;
    float d0 = 0.0f, d1 = 0.0f;
    for (int j = lane; j < Cc; j += 32) {
        float v = r[j];
        d0 += v * w2[j];
        d1 += v * w2[Cc + j];
    }
    for (int o = 16; o > 0; o >>= 1) { d0 += __shfl_xor(d0, o); d1 += __shfl_xor(d1, o); }
    if (lane == 0) {
        float del[2] = {d0 + b2[0], d1 + b2[1]};
        int b = wid / Kq, k = wid % Kq;
        #pragma unroll
        for (int c = 0; c < 2; ++c) {
            float x = fminf(fmaxf(pts[wid * 2 + c], 0.0f), 1.0f);
            float is = logf(fmaxf(x, 1e-5f) / fmaxf(1.0f - x, 1e-5f));
            float rp = 1.0f / (1.0f + expf(-(is + del[c])));
            pts[wid * 2 + c] = rp;
            out_pts[(((size_t)b * LOUTn + layer) * Kq + k) * 2 + c] = rp;
        }
    }
}

// ============================================================
// host helpers
// ============================================================
static inline void launch_gemm(const __bf16* A, const __bf16* W, const float* bias, float* C,
                               int M, int Kd, int N, int relu, hipStream_t s)
{
    int waves  = (M / 32) * (N / 64);
    int blocks = (waves + 7) / 8;
    wmma_gemm_bf16<<<blocks, 256, 0, s>>>(A, W, bias, C, M, Kd, N, relu);
}

static inline void launch_cvt(const float* a, __bf16* o, size_t n, hipStream_t s)
{
    int blocks = (int)((n + 255) / 256);
    cvt_bf16_kernel<<<blocks, 256, 0, s>>>(a, o, n);
}

extern "C" void kernel_launch(void* const* d_in, const int* in_sizes, int n_in,
                              void* d_out, int out_size, void* d_ws, size_t ws_size,
                              hipStream_t stream)
{
    (void)in_sizes; (void)n_in; (void)out_size; (void)ws_size;

    const float* query_embed = (const float*)d_in[0];
    const float* feat0 = (const float*)d_in[1];
    const float* feat1 = (const float*)d_in[2];
    const float* feat2 = (const float*)d_in[3];
    const float* feat3 = (const float*)d_in[4];
    const float* init_points = (const float*)d_in[5];
    const float* sa_in_w  = (const float*)d_in[6];
    const float* sa_in_b  = (const float*)d_in[7];
    const float* sa_out_w = (const float*)d_in[8];
    const float* sa_out_b = (const float*)d_in[9];
    const float* sa_ln_g  = (const float*)d_in[10];
    const float* sa_ln_b  = (const float*)d_in[11];
    const float* off_w = (const float*)d_in[12];
    const float* off_b = (const float*)d_in[13];
    const float* aw_w  = (const float*)d_in[14];
    const float* aw_b  = (const float*)d_in[15];
    const float* vp_w  = (const float*)d_in[16];
    const float* vp_b  = (const float*)d_in[17];
    const float* op_w  = (const float*)d_in[18];
    const float* op_b  = (const float*)d_in[19];
    const float* ln1_g = (const float*)d_in[20];
    const float* ln1_b = (const float*)d_in[21];
    const float* ff1_w = (const float*)d_in[22];
    const float* ff1_b = (const float*)d_in[23];
    const float* ff2_w = (const float*)d_in[24];
    const float* ff2_b = (const float*)d_in[25];
    const float* ln2_g = (const float*)d_in[26];
    const float* ln2_b = (const float*)d_in[27];
    const float* ph_w1 = (const float*)d_in[28];
    const float* ph_b1 = (const float*)d_in[29];
    const float* ph_w2 = (const float*)d_in[30];
    const float* ph_b2 = (const float*)d_in[31];

    // ---- workspace carve-up (float units, every chunk 16B aligned) ----
    float* ws = (float*)d_ws;
    size_t cur = 0;
    auto allocf  = [&](size_t n) { float* p = ws + cur; cur += n; return p; };
    auto allocbf = [&](size_t n) { __bf16* p = (__bf16*)(ws + cur); cur += n / 2; return p; };
    const size_t VN = (size_t)Bn * Ntot * Cc;   // 44,564,480

    float*  value    = allocf(VN);
    float*  vproj    = allocf(VN);
    __bf16* value_bf = allocbf(VN);
    float*  qe    = allocf(BKC);
    float*  qpos  = allocf(BKC);
    float*  q2    = allocf(BKC);
    float*  qb    = allocf(BKC);
    float*  kb    = allocf(BKC);
    float*  vb    = allocf(BKC);
    float*  ob    = allocf(BKC);
    float*  t2    = allocf(BKC);
    float*  offb  = allocf(BKC);
    float*  awb   = allocf((size_t)BK * 128);
    float*  dob   = allocf(BKC);
    float*  t3    = allocf(BKC);
    float*  xb    = allocf(BKC);
    float*  ffh   = allocf((size_t)BK * DFFn);
    float*  fout  = allocf(BKC);
    float*  phh   = allocf(BKC);
    float*  validb = allocf(BK);
    float*  pts   = allocf((size_t)BK * 2);
    // bf16 activation scratch
    __bf16* abf0 = allocbf(BKC);                  // q2 mirror
    __bf16* abf1 = allocbf((size_t)BK * DFFn);    // rotating operand buffer
    // bf16 weights (all layers, converted once)
    __bf16* sa_in_w_bf  = allocbf((size_t)LOUTn * 3 * Cc * Cc);
    __bf16* sa_out_w_bf = allocbf((size_t)LOUTn * Cc * Cc);
    __bf16* off_w_bf    = allocbf((size_t)LOUTn * Cc * Cc);
    __bf16* aw_w_bf     = allocbf((size_t)LOUTn * 128 * Cc);
    __bf16* vp_w_bf     = allocbf((size_t)LOUTn * Cc * Cc);
    __bf16* op_w_bf     = allocbf((size_t)LOUTn * Cc * Cc);
    __bf16* ff1_w_bf    = allocbf((size_t)LOUTn * DFFn * Cc);
    __bf16* ff2_w_bf    = allocbf((size_t)LOUTn * Cc * DFFn);
    __bf16* ph_w1_bf    = allocbf((size_t)LOUTn * Cc * Cc);

    float* out_refined = (float*)d_out;
    float* out_pts     = (float*)d_out + BKC;

    // ---- one-time setup ----
    hipMemcpyAsync(qe,  query_embed, BKC * sizeof(float), hipMemcpyDeviceToDevice, stream);
    hipMemcpyAsync(pts, init_points, (size_t)BK * 2 * sizeof(float), hipMemcpyDeviceToDevice, stream);
    build_value_kernel<<<(int)((VN + 255) / 256), 256, 0, stream>>>(
        feat0, feat1, feat2, feat3, value, value_bf);
    valid_kernel<<<(BK + 7) / 8, 256, 0, stream>>>(query_embed, validb);

    launch_cvt(sa_in_w,  sa_in_w_bf,  (size_t)LOUTn * 3 * Cc * Cc, stream);
    launch_cvt(sa_out_w, sa_out_w_bf, (size_t)LOUTn * Cc * Cc, stream);
    launch_cvt(off_w,    off_w_bf,    (size_t)LOUTn * Cc * Cc, stream);
    launch_cvt(aw_w,     aw_w_bf,     (size_t)LOUTn * 128 * Cc, stream);
    launch_cvt(vp_w,     vp_w_bf,     (size_t)LOUTn * Cc * Cc, stream);
    launch_cvt(op_w,     op_w_bf,     (size_t)LOUTn * Cc * Cc, stream);
    launch_cvt(ff1_w,    ff1_w_bf,    (size_t)LOUTn * DFFn * Cc, stream);
    launch_cvt(ff2_w,    ff2_w_bf,    (size_t)LOUTn * Cc * DFFn, stream);
    launch_cvt(ph_w1,    ph_w1_bf,    (size_t)LOUTn * Cc * Cc, stream);

    const int nElem = (int)BKC;                 // 819200
    const int eBlocks = nElem / 256;            // 3200

    for (int l = 0; l < LOUTn; ++l) {
        const __bf16* sw  = sa_in_w_bf  + (size_t)l * 3 * Cc * Cc;
        const float*  sb  = sa_in_b     + (size_t)l * 3 * Cc;
        const __bf16* sow = sa_out_w_bf + (size_t)l * Cc * Cc;
        const float*  sob = sa_out_b    + (size_t)l * Cc;
        const float*  slg = sa_ln_g     + (size_t)l * Cc;
        const float*  slb = sa_ln_b     + (size_t)l * Cc;
        const __bf16* owl = off_w_bf + (size_t)l * Cc * Cc;
        const float*  obl = off_b    + (size_t)l * Cc;
        const __bf16* awl = aw_w_bf  + (size_t)l * 128 * Cc;
        const float*  abl = aw_b     + (size_t)l * 128;
        const __bf16* vwl = vp_w_bf  + (size_t)l * Cc * Cc;
        const float*  vbl = vp_b     + (size_t)l * Cc;
        const __bf16* opl = op_w_bf  + (size_t)l * Cc * Cc;
        const float*  opb = op_b     + (size_t)l * Cc;
        const float*  l1g = ln1_g + (size_t)l * Cc;
        const float*  l1b = ln1_b + (size_t)l * Cc;
        const __bf16* f1w = ff1_w_bf + (size_t)l * DFFn * Cc;
        const float*  f1b = ff1_b    + (size_t)l * DFFn;
        const __bf16* f2w = ff2_w_bf + (size_t)l * Cc * DFFn;
        const float*  f2b = ff2_b    + (size_t)l * Cc;
        const float*  l2g = ln2_g + (size_t)l * Cc;
        const float*  l2b = ln2_b + (size_t)l * Cc;
        const __bf16* p1w = ph_w1_bf + (size_t)l * Cc * Cc;
        const float*  p1b = ph_b1    + (size_t)l * Cc;
        const float*  p2w = ph_w2 + (size_t)l * 2 * Cc;
        const float*  p2b = ph_b2 + (size_t)l * 2;

        // positional encoding of current points
        coord_pos_kernel<<<eBlocks, 256, 0, stream>>>(pts, qpos);

        // ---- self attention ----
        add_vec_kernel<<<eBlocks, 256, 0, stream>>>(qe, qpos, q2, abf0, nElem);   // q2 (+ bf16)
        launch_cvt(qe, abf1, BKC, stream);                                        // qe bf16 (V input)
        launch_gemm(abf0, sw,                   sb,          qb, BK, Cc, Cc, 0, stream); // Q
        launch_gemm(abf0, sw + (size_t)Cc*Cc,   sb + Cc,     kb, BK, Cc, Cc, 0, stream); // K
        launch_gemm(abf1, sw + (size_t)2*Cc*Cc, sb + 2*Cc,   vb, BK, Cc, Cc, 0, stream); // V
        sa_attn_kernel<<<(Bn*HHn*Kq) / 8, 256, 0, stream>>>(qb, kb, vb, ob);
        launch_cvt(ob, abf1, BKC, stream);
        launch_gemm(abf1, sow, sob, t2, BK, Cc, Cc, 0, stream);
        add_ln_kernel<<<(BK + 7) / 8, 256, 0, stream>>>(qe, t2, slg, slb, validb, qe, nullptr, 1);

        // ---- reference-feature gather ----
        ref_feat_kernel<<<BK, 256, 0, stream>>>(value, pts, validb, qe);

        // ---- deformable attention ----
        add_vec_kernel<<<eBlocks, 256, 0, stream>>>(qe, qpos, q2, abf0, nElem);
        launch_gemm(abf0, owl, obl, offb, BK, Cc, Cc, 0, stream);        // offsets [.,256]
        launch_gemm(abf0, awl, abl, awb,  BK, Cc, 128, 0, stream);       // attn weights [.,128]
        launch_gemm(value_bf, vwl, vbl, vproj, Bn * Ntot, Cc, Cc, 0, stream); // value proj (big)
        deform_kernel<<<(Bn*Kq*HHn) / 8, 256, 0, stream>>>(vproj, offb, awb, pts, dob, abf1);
        launch_gemm(abf1, opl, opb, t3, BK, Cc, Cc, 0, stream);
        add_ln_kernel<<<(BK + 7) / 8, 256, 0, stream>>>(qe, t3, l1g, l1b, validb, xb, abf1, 0);

        // ---- FFN ----
        launch_gemm(abf1, f1w, f1b, ffh, BK, Cc, DFFn, 1, stream);       // relu
        launch_cvt(ffh, abf1, (size_t)BK * DFFn, stream);
        launch_gemm(abf1, f2w, f2b, fout, BK, DFFn, Cc, 0, stream);
        add_ln_kernel<<<(BK + 7) / 8, 256, 0, stream>>>(xb, fout, l2g, l2b, validb, qe, abf1, 1); // qe = refined

        // ---- point head + refinement ----
        launch_gemm(abf1, p1w, p1b, phh, BK, Cc, Cc, 1, stream);         // relu
        point_head_kernel<<<(BK + 7) / 8, 256, 0, stream>>>(phh, p2w, p2b, pts, out_pts, l);
    }

    hipMemcpyAsync(out_refined, qe, BKC * sizeof(float), hipMemcpyDeviceToDevice, stream);
}
